// TinyMambaLM_13649406067285
// MI455X (gfx1250) — matrix operand order repaired and checked by reading, not compiled, not run on hardware
//
#include <hip/hip_runtime.h>
#include <hip/hip_bf16.h>
#include <math.h>

// ---------------- model constants ----------------
#define V_    32000
#define DM_   512
#define NL_   4
#define DS_   16
#define DC_   4
#define DI_   1024          // EXP * DM
#define DTR_  32            // DM / 16
#define B_    2
#define L_    512
#define T_    (B_ * L_)     // 1024 token rows
#define XPN_  (DTR_ + 2 * DS_)   // 64
#define EPS_  1e-5f

typedef __attribute__((ext_vector_type(16))) __bf16 v16bf;
typedef __attribute__((ext_vector_type(8)))  __bf16 v8bf;
typedef __attribute__((ext_vector_type(8)))  float  v8f;
typedef __attribute__((ext_vector_type(4)))  int    v4i;

#define AS1 __attribute__((address_space(1)))
#define AS3 __attribute__((address_space(3)))

#if __has_builtin(__builtin_amdgcn_global_load_async_to_lds_b128)
#define HAS_ASYNC_LDS 1
// signature (probe-confirmed by clang diagnostic): (v4i AS1*, v4i AS3*, imm offset, imm cpol)
#define ASYNC_CP_B128(srcp, dstp) \
  __builtin_amdgcn_global_load_async_to_lds_b128((AS1 v4i*)(srcp), (AS3 v4i*)(dstp), 0, 0)
#else
#define HAS_ASYNC_LDS 0
#endif

__device__ __forceinline__ void wait_async_lds() {
#if HAS_ASYNC_LDS
#if __has_builtin(__builtin_amdgcn_s_wait_asynccnt)
  __builtin_amdgcn_s_wait_asynccnt(0);
#else
  asm volatile("s_wait_asynccnt 0" ::: "memory");
#endif
#endif
}

// ---------------- elementwise helpers ----------------
__global__ void embed_kernel(const int* __restrict__ idx,
                             const float* __restrict__ emb,
                             float* __restrict__ x) {
  int t = blockIdx.x;                      // 0..T_-1
  int tok = idx[t];
  for (int d = threadIdx.x; d < DM_; d += blockDim.x)
    x[t * DM_ + d] = emb[tok * DM_ + d];
}

// LayerNorm over DM_=512, one block per row, bf16 output for WMMA consumption.
__global__ void layernorm_bf16_kernel(const float* __restrict__ x,
                                      const float* __restrict__ w,
                                      const float* __restrict__ b,
                                      __bf16* __restrict__ out) {
  __shared__ float red[16];
  int t = blockIdx.x;
  const float* xr = x + t * DM_;
  float s = 0.f, ss = 0.f;
  for (int d = threadIdx.x; d < DM_; d += blockDim.x) {
    float v = xr[d]; s += v; ss += v * v;
  }
  for (int o = 16; o > 0; o >>= 1) {
    s  += __shfl_down(s,  o, 32);
    ss += __shfl_down(ss, o, 32);
  }
  int wid = threadIdx.x >> 5, lane = threadIdx.x & 31;
  if (lane == 0) { red[wid] = s; red[8 + wid] = ss; }
  __syncthreads();
  if (threadIdx.x == 0) {
    float a = 0.f, c = 0.f;
    int nw = blockDim.x >> 5;
    for (int i = 0; i < nw; i++) { a += red[i]; c += red[8 + i]; }
    red[0] = a; red[8] = c;
  }
  __syncthreads();
  float mean = red[0] / (float)DM_;
  float var  = red[8] / (float)DM_ - mean * mean;
  float rstd = rsqrtf(var + EPS_);
  for (int d = threadIdx.x; d < DM_; d += blockDim.x) {
    float v = (xr[d] - mean) * rstd * w[d] + b[d];
    out[t * DM_ + d] = (__bf16)v;
  }
}

// ---------------- weight pack: f32 [K x N] -> bf16 WMMA-fragment order ------
// Fragment (kt, nt) covers K rows [kt*32, kt*32+32) x N cols [nt*16, nt*16+16).
// Linear fragment id = nt * (K/32) + kt; 512 bf16 per fragment, stored so that
// lane l owns 16 contiguous elements: elem e = B[kt*32 + (l>>4)*16 + e][nt*16 + (l&15)].
__global__ void pack_b_kernel(const float* __restrict__ B,
                              __bf16* __restrict__ Bp, int K, int N) {
  int i = blockIdx.x * blockDim.x + threadIdx.x;
  if (i >= K * N) return;
  int frag = i >> 9;
  int rem  = i & 511;
  int lane = rem >> 4;
  int e    = rem & 15;
  int ktiles = K >> 5;
  int nt = frag / ktiles;
  int kt = frag - nt * ktiles;
  int k = (kt << 5) + ((lane >> 4) << 4) + e;
  int n = (nt << 4) + (lane & 15);
  Bp[i] = (__bf16)B[(size_t)k * N + n];
}

// ---------------- WMMA bf16 GEMM ----------------
// A: [M x K] row-major bf16 (staged through LDS via async copy, double-buffered)
// Bp: fragment-packed bf16 (see pack_b_kernel)
// C: [M x N] f32 (ACC=1 -> fused residual accumulate)
// Block = 128 thr = 4 waves (2x2), block tile 64x64; wave tile 32x32 = 2x2 WMMA.
// M, N multiples of 64; K multiple of 32.

__device__ __forceinline__ v16bf load_a_frag_lds(const __bf16* sm, int r0) {
  int lane = threadIdx.x & 31;
  const __bf16* p = sm + (r0 + (lane & 15)) * 32 + ((lane >> 4) << 4);
  v8bf lo = *(const v8bf*)p;          // K = kb .. kb+7      (ds_load_b128)
  v8bf hi = *(const v8bf*)(p + 8);    // K = kb+8 .. kb+15   (ds_load_b128)
  return __builtin_shufflevector(lo, hi, 0, 1, 2, 3, 4, 5, 6, 7,
                                         8, 9, 10, 11, 12, 13, 14, 15);
}

template <int ACC>
__device__ __forceinline__ void store_c_frag(float* __restrict__ C, int ldc,
                                             int row0, int col0, v8f c) {
  int lane = threadIdx.x & 31;
  float* p = C + (size_t)(row0 + ((lane >> 4) << 3)) * ldc + col0 + (lane & 15);
#pragma unroll
  for (int v = 0; v < 8; v++) {
    if (ACC) *p += c[v]; else *p = c[v];
    p += ldc;
  }
}

template <int ACC>
__global__ void gemm_bf16_wmma(const __bf16* __restrict__ A,
                               const __bf16* __restrict__ Bp,
                               float* __restrict__ C,
                               int M, int N, int K) {
  __shared__ __align__(32) __bf16 smA[2][64 * 32];
  int tid = threadIdx.x;
  int lane = tid & 31;
  int wid = tid >> 5;
  int wm = wid & 1, wn = wid >> 1;
  int rowB = blockIdx.y * 64;
  int row  = rowB + wm * 32;
  int col  = blockIdx.x * 64 + wn * 32;
  int nk = K >> 5;

  // per-thread A staging addresses: two 16B chunks per 64x32 tile
  int r0 = tid >> 2;                   // 0..31
  int kk = (tid & 3) << 3;             // 0,8,16,24
  const __bf16* ga0 = A + (size_t)(rowB + r0) * K + kk;
  const __bf16* ga1 = A + (size_t)(rowB + 32 + r0) * K + kk;
  int lofs = r0 * 32 + kk;

  // B fragment pointers (packed): advance 512 elements per K-step
  int ct = col >> 4;
  const __bf16* pb0 = Bp + (((size_t)ct * nk) << 9) + (lane << 4);
  const __bf16* pb1 = pb0 + (((size_t)nk) << 9);

  v8f c00 = {}, c01 = {}, c10 = {}, c11 = {};

  // prologue: stage K-tile 0 into buffer 0
#if HAS_ASYNC_LDS
  ASYNC_CP_B128(ga0, &smA[0][lofs]);
  ASYNC_CP_B128(ga1, &smA[0][32 * 32 + lofs]);
#else
  *(v8bf*)(&smA[0][lofs])           = *(const v8bf*)ga0;
  *(v8bf*)(&smA[0][32 * 32 + lofs]) = *(const v8bf*)ga1;
#endif
  ga0 += 32; ga1 += 32;

  for (int kt = 0; kt < nk; kt++) {
    wait_async_lds();
    __syncthreads();
    if (kt + 1 < nk) {                  // stage next tile into other buffer
      __bf16* dst = smA[(kt + 1) & 1];
#if HAS_ASYNC_LDS
      ASYNC_CP_B128(ga0, dst + lofs);
      ASYNC_CP_B128(ga1, dst + 32 * 32 + lofs);
#else
      *(v8bf*)(dst + lofs)           = *(const v8bf*)ga0;
      *(v8bf*)(dst + 32 * 32 + lofs) = *(const v8bf*)ga1;
#endif
      ga0 += 32; ga1 += 32;
    }
    const __bf16* sm = smA[kt & 1];
    v16bf a0 = load_a_frag_lds(sm, wm * 32);
    v16bf a1 = load_a_frag_lds(sm, wm * 32 + 16);
    v16bf b0 = *(const v16bf*)pb0;      // 2x global_load_b128, contiguous
    v16bf b1 = *(const v16bf*)pb1;
    __builtin_prefetch(pb0 + 512, 0, 0);   // next K-step B fragments
    __builtin_prefetch(pb1 + 512, 0, 0);
    pb0 += 512; pb1 += 512;
    c00 = __builtin_amdgcn_wmma_f32_16x16x32_bf16(false, a0, false, b0, (short)0, c00, false, false);
    c01 = __builtin_amdgcn_wmma_f32_16x16x32_bf16(false, a0, false, b1, (short)0, c01, false, false);
    c10 = __builtin_amdgcn_wmma_f32_16x16x32_bf16(false, a1, false, b0, (short)0, c10, false, false);
    c11 = __builtin_amdgcn_wmma_f32_16x16x32_bf16(false, a1, false, b1, (short)0, c11, false, false);
  }
  store_c_frag<ACC>(C, N, row,      col,      c00);
  store_c_frag<ACC>(C, N, row,      col + 16, c01);
  store_c_frag<ACC>(C, N, row + 16, col,      c10);
  store_c_frag<ACC>(C, N, row + 16, col + 16, c11);
}

// ---------------- depthwise causal conv + SiLU ----------------
__global__ void conv_silu_kernel(const float* __restrict__ xz,   // [T, 2*DI]
                                 const float* __restrict__ cw,   // [DI, DC]
                                 const float* __restrict__ cb,   // [DI]
                                 float* __restrict__ xc,
                                 __bf16* __restrict__ xcb) {
  int i = blockIdx.x * blockDim.x + threadIdx.x;   // over T_*DI_
  if (i >= T_ * DI_) return;
  int d = i % DI_;
  int t = i / DI_;
  int l = t % L_;
  float acc = cb[d];
#pragma unroll
  for (int k = 0; k < DC_; k++) {
    int ll = l - (DC_ - 1) + k;
    if (ll >= 0)
      acc += xz[(size_t)(t - l + ll) * (2 * DI_) + d] * cw[d * DC_ + k];
  }
  float sv = acc / (1.f + __expf(-acc));   // SiLU
  xc[i]  = sv;
  xcb[i] = (__bf16)sv;
}

// dt slice of x_dbl [T, 64] -> bf16 [T, 32]
__global__ void dt_extract_kernel(const float* __restrict__ xdbl,
                                  __bf16* __restrict__ dtb) {
  int i = blockIdx.x * blockDim.x + threadIdx.x;   // T_*DTR_
  if (i >= T_ * DTR_) return;
  int t = i / DTR_, c = i % DTR_;
  dtb[i] = (__bf16)xdbl[t * XPN_ + c];
}

// delta = softplus(delta_raw + dt_b[d]), in place on [T, DI]
__global__ void softplus_kernel(float* __restrict__ delta,
                                const float* __restrict__ dtb) {
  int i = blockIdx.x * blockDim.x + threadIdx.x;
  if (i >= T_ * DI_) return;
  int d = i % DI_;
  float x = delta[i] + dtb[d];
  delta[i] = (x > 20.f) ? x : log1pf(__expf(x));
}

// Selective scan + D skip + SiLU(res) gate, one thread per (b, d) channel.
__global__ void scan_gate_kernel(const float* __restrict__ xc,     // [T, DI] u
                                 const float* __restrict__ delta,  // [T, DI]
                                 const float* __restrict__ xdbl,   // [T, 64]
                                 const float* __restrict__ A_log,  // [DI, DS]
                                 const float* __restrict__ Dp,     // [DI]
                                 const float* __restrict__ xz,     // [T, 2*DI]
                                 __bf16* __restrict__ yb) {        // [T, DI]
  int i = blockIdx.x * blockDim.x + threadIdx.x;   // B_*DI_
  if (i >= B_ * DI_) return;
  int b = i / DI_, d = i % DI_;
  float Av[DS_], h[DS_];
#pragma unroll
  for (int n = 0; n < DS_; n++) {
    Av[n] = -__expf(A_log[d * DS_ + n]);
    h[n]  = 0.f;
  }
  float Dd = Dp[d];
  for (int l = 0; l < L_; l++) {
    int t = b * L_ + l;
    float dt = delta[(size_t)t * DI_ + d];
    float u  = xc[(size_t)t * DI_ + d];
    const float* xr = xdbl + (size_t)t * XPN_;
    float dtu = dt * u;
    float acc = 0.f;
#pragma unroll
    for (int n = 0; n < DS_; n++) {
      float dA = __expf(dt * Av[n]);
      h[n] = dA * h[n] + dtu * xr[DTR_ + n];        // B_t[n]
      acc += h[n] * xr[DTR_ + DS_ + n];             // C_t[n]
    }
    float y = acc + u * Dd;
    float r = xz[(size_t)t * (2 * DI_) + DI_ + d];
    y *= r / (1.f + __expf(-r));                    // * SiLU(res)
    yb[(size_t)t * DI_ + d] = (__bf16)y;
  }
}

// ---------------- host-side orchestration ----------------
static inline size_t align_up(size_t v) { return (v + 255) & ~(size_t)255; }

extern "C" void kernel_launch(void* const* d_in, const int* in_sizes, int n_in,
                              void* d_out, int out_size, void* d_ws, size_t ws_size,
                              hipStream_t stream) {
  (void)in_sizes; (void)n_in; (void)out_size; (void)ws_size;

  const int*   idx      = (const int*)  d_in[0];
  const float* tok_emb  = (const float*)d_in[1];
  const float* ln_w     = (const float*)d_in[2];
  const float* ln_b     = (const float*)d_in[3];
  const float* in_proj  = (const float*)d_in[4];
  const float* conv_w   = (const float*)d_in[5];
  const float* conv_b   = (const float*)d_in[6];
  const float* x_proj   = (const float*)d_in[7];
  const float* dt_w     = (const float*)d_in[8];
  const float* dt_b     = (const float*)d_in[9];
  const float* A_log    = (const float*)d_in[10];
  const float* Dparam   = (const float*)d_in[11];
  const float* out_proj = (const float*)d_in[12];
  const float* lnf_w    = (const float*)d_in[13];
  const float* lnf_b    = (const float*)d_in[14];
  const float* head_w   = (const float*)d_in[15];
  float* out = (float*)d_out;

  // workspace carve-up
  char* ws = (char*)d_ws;
  size_t o = 0;
  float*  x_f32   = (float*) (ws + o); o = align_up(o + (size_t)T_ * DM_ * 4);
  __bf16* h_bf    = (__bf16*)(ws + o); o = align_up(o + (size_t)T_ * DM_ * 2);
  float*  xz      = (float*) (ws + o); o = align_up(o + (size_t)T_ * 2 * DI_ * 4);
  float*  xc_f32  = (float*) (ws + o); o = align_up(o + (size_t)T_ * DI_ * 4);
  __bf16* xc_bf   = (__bf16*)(ws + o); o = align_up(o + (size_t)T_ * DI_ * 2);
  float*  x_dbl   = (float*) (ws + o); o = align_up(o + (size_t)T_ * XPN_ * 4);
  __bf16* dt_bf   = (__bf16*)(ws + o); o = align_up(o + (size_t)T_ * DTR_ * 2);
  float*  delta   = (float*) (ws + o); o = align_up(o + (size_t)T_ * DI_ * 4);
  __bf16* y_bf    = (__bf16*)(ws + o); o = align_up(o + (size_t)T_ * DI_ * 2);
  __bf16* w_in    = (__bf16*)(ws + o); o = align_up(o + (size_t)NL_ * DM_ * 2 * DI_ * 2);
  __bf16* w_xp    = (__bf16*)(ws + o); o = align_up(o + (size_t)NL_ * DI_ * XPN_ * 2);
  __bf16* w_dtw   = (__bf16*)(ws + o); o = align_up(o + (size_t)NL_ * DTR_ * DI_ * 2);
  __bf16* w_out   = (__bf16*)(ws + o); o = align_up(o + (size_t)NL_ * DI_ * DM_ * 2);
  __bf16* w_head  = (__bf16*)(ws + o); o = align_up(o + (size_t)DM_ * V_ * 2);

  const int CT = 256;
  auto nb = [](int n, int b) { return (n + b - 1) / b; };

  // pack weights into WMMA fragment order (bf16), every call (deterministic)
  for (int l = 0; l < NL_; l++) {
    { int n = DM_ * 2 * DI_; pack_b_kernel<<<nb(n, CT), CT, 0, stream>>>(
          in_proj + (size_t)l * n, w_in + (size_t)l * n, DM_, 2 * DI_); }
    { int n = DI_ * XPN_;    pack_b_kernel<<<nb(n, CT), CT, 0, stream>>>(
          x_proj + (size_t)l * n, w_xp + (size_t)l * n, DI_, XPN_); }
    { int n = DTR_ * DI_;    pack_b_kernel<<<nb(n, CT), CT, 0, stream>>>(
          dt_w + (size_t)l * n, w_dtw + (size_t)l * n, DTR_, DI_); }
    { int n = DI_ * DM_;     pack_b_kernel<<<nb(n, CT), CT, 0, stream>>>(
          out_proj + (size_t)l * n, w_out + (size_t)l * n, DI_, DM_); }
  }
  { int n = DM_ * V_; pack_b_kernel<<<nb(n, CT), CT, 0, stream>>>(head_w, w_head, DM_, V_); }

  // embedding gather
  embed_kernel<<<T_, 256, 0, stream>>>(idx, tok_emb, x_f32);

  for (int i = 0; i < NL_; i++) {
    layernorm_bf16_kernel<<<T_, 256, 0, stream>>>(x_f32, ln_w + i * DM_, ln_b + i * DM_, h_bf);
    // xz = h @ in_proj   [1024x512]·[512x2048]
    gemm_bf16_wmma<0><<<dim3((2 * DI_) / 64, T_ / 64), 128, 0, stream>>>(
        h_bf, w_in + (size_t)i * DM_ * 2 * DI_, xz, T_, 2 * DI_, DM_);
    conv_silu_kernel<<<nb(T_ * DI_, CT), CT, 0, stream>>>(
        xz, conv_w + (size_t)i * DI_ * DC_, conv_b + (size_t)i * DI_, xc_f32, xc_bf);
    // x_dbl = xc @ x_proj   [1024x1024]·[1024x64]
    gemm_bf16_wmma<0><<<dim3(XPN_ / 64, T_ / 64), 128, 0, stream>>>(
        xc_bf, w_xp + (size_t)i * DI_ * XPN_, x_dbl, T_, XPN_, DI_);
    dt_extract_kernel<<<nb(T_ * DTR_, CT), CT, 0, stream>>>(x_dbl, dt_bf);
    // delta_raw = dt @ dt_w   [1024x32]·[32x1024]
    gemm_bf16_wmma<0><<<dim3(DI_ / 64, T_ / 64), 128, 0, stream>>>(
        dt_bf, w_dtw + (size_t)i * DTR_ * DI_, delta, T_, DI_, DTR_);
    softplus_kernel<<<nb(T_ * DI_, CT), CT, 0, stream>>>(delta, dt_b + (size_t)i * DI_);
    scan_gate_kernel<<<nb(B_ * DI_, CT), CT, 0, stream>>>(
        xc_f32, delta, x_dbl, A_log + (size_t)i * DI_ * DS_,
        Dparam + (size_t)i * DI_, xz, y_bf);
    // x += y @ out_proj   [1024x1024]·[1024x512], fused residual add
    gemm_bf16_wmma<1><<<dim3(DM_ / 64, T_ / 64), 128, 0, stream>>>(
        y_bf, w_out + (size_t)i * DI_ * DM_, x_f32, T_, DM_, DI_);
  }

  // final LN -> bf16, then logits = x @ head_w  [1024x512]·[512x32000]
  layernorm_bf16_kernel<<<T_, 256, 0, stream>>>(x_f32, lnf_w, lnf_b, h_bf);
  gemm_bf16_wmma<0><<<dim3(V_ / 64, T_ / 64), 128, 0, stream>>>(
      h_bf, w_head, out, T_, V_, DM_);
}
